// LOBModel_69861938037094
// MI455X (gfx1250) — matrix operand order repaired and checked by reading, not compiled, not run on hardware
//
#include <hip/hip_runtime.h>
#include <cstdint>

#define BB 8
#define LL 2048
#define FF 40
#define EE 128
#define NLAYERS 4
#define DSTATE 16
#define DCONV 4
#define DINNER 256
#define DTRANK 8
#define BL (BB*LL)   /* 16384 rows */

typedef __attribute__((ext_vector_type(16))) _Float16 v16h;
typedef __attribute__((ext_vector_type(8)))  _Float16 v8h;
typedef __attribute__((ext_vector_type(2)))  _Float16 v2h;
typedef __attribute__((ext_vector_type(8)))  float    v8f;

__device__ __forceinline__ float geluf(float x) {
  return 0.5f * x * (1.0f + erff(x * 0.70710678118654752f));
}
__device__ __forceinline__ float siluf(float x) {
  return x / (1.0f + __expf(-x));
}

// ---------------------------------------------------------------------------
// Generic WMMA GEMM:  C[M,N] (+)= (A[M,K] * B[K,N]) (+bias)
// kind 0: A row-major (lda), B row-major (K,N)
// kind 1: conv k=3, pad=1 : A = im2col from X laid out (B,L,Cin), lda=Cin,
//         K = 3*Cin, k -> (t = k/Cin, ci = k%Cin), reads X[b, l+t-1, ci];
//         B from W(Cout,Cin,3): B[k][n] = W[n*Cin*3 + ci*3 + t]
// kind 2: conv k=1 shortcut: A plain (BL,Cin); B from W(Cout,Cin): W[n*Cin+k]
// flags bit0: add bias[n];  bit1: accumulate into existing C (residual)
// Tile: 64 x 128, BK=32, 4 waves; LDS staged so every fragment is 2x b128;
// all 8 B fragments prefetched to registers, then 8 WMMAs issue back-to-back.
// ---------------------------------------------------------------------------
#define BM 64
#define BN 128
#define BKK 32
#define NT (BN/16)    /* 8 WMMA tiles per wave per K-step */
#define LDK (BKK + 8) /* 40 halves = 80B row stride (16B aligned) */

__global__ __launch_bounds__(128)
void gemm_wmma_f16(const float* __restrict__ A, int lda,
                   const float* __restrict__ Bw,
                   const float* __restrict__ bias,
                   float* __restrict__ C, int ldc,
                   int M, int N, int K,
                   int kind, int flags, int Cin, int Lseq)
{
  __shared__ _Float16 sA [BM][LDK];   // row-major A tile
  __shared__ _Float16 sBt[BN][LDK];   // TRANSPOSED B tile: [col][k]

  const int tid  = threadIdx.x;      // 0..127
  const int wave = tid >> 5;         // 0..3  (wave32)
  const int lane = tid & 31;
  const int m0 = blockIdx.y * BM;
  const int n0 = blockIdx.x * BN;

  v8f acc[NT];
  #pragma unroll
  for (int i = 0; i < NT; ++i)
    #pragma unroll
    for (int j = 0; j < 8; ++j) acc[i][j] = 0.0f;

  for (int k0 = 0; k0 < K; k0 += BKK) {
    // ---- stage A tile (64x32) as K-pairs -> packed v2h stores ----
    for (int i = tid; i < BM * BKK / 2; i += 128) {
      int r  = i >> 4;        // row in tile 0..63
      int kp = i & 15;        // K-pair index
      int m  = m0 + r;
      int k  = k0 + 2 * kp;
      float va = 0.0f, vb = 0.0f;
      if (m < M) {
        if (kind == 1) {
          int b  = m / Lseq;
          int l  = m - b * Lseq;
          int t  = k / Cin;
          int ci = k - t * Cin;        // ci even; ci+1 < Cin guaranteed
          int ll = l + t - 1;          // pad = 1
          if (ll >= 0 && ll < Lseq) {
            size_t base = (size_t)(b * Lseq + ll) * lda + ci;
            va = A[base]; vb = A[base + 1];
          }
        } else {
          size_t base = (size_t)m * lda + k;
          if (k     < K) va = A[base];
          if (k + 1 < K) vb = A[base + 1];
        }
      }
      v2h p; p[0] = (_Float16)va; p[1] = (_Float16)vb;
      *(v2h*)&sA[r][2 * kp] = p;
    }
    // ---- stage B tile (32x128) transposed, as K-pairs ----
    for (int i = tid; i < BKK * BN / 2; i += 128) {
      int c  = i & (BN - 1);  // column 0..127 (coalesced along n)
      int kp = i >> 7;        // K-pair index 0..15
      int k  = k0 + 2 * kp;
      int n  = n0 + c;
      float va = 0.0f, vb = 0.0f;
      if (n < N) {
        if (kind == 0) {
          if (k     < K) va = Bw[(size_t)k * N + n];
          if (k + 1 < K) vb = Bw[(size_t)(k + 1) * N + n];
        } else if (kind == 1) {
          int t  = k / Cin;
          int ci = k - t * Cin;
          size_t base = ((size_t)n * Cin + ci) * 3 + t;
          va = Bw[base]; vb = Bw[base + 3];   // (ci+1) stays in-bounds
        } else {
          size_t base = (size_t)n * Cin + k;
          if (k     < K) va = Bw[base];
          if (k + 1 < K) vb = Bw[base + 1];
        }
      }
      v2h p; p[0] = (_Float16)va; p[1] = (_Float16)vb;
      *(v2h*)&sBt[c][2 * kp] = p;
    }
    __syncthreads();

    // ---- fragments: each is two contiguous 16B LDS reads (ds_load_b128).
    //      Prefetch ALL B fragments, then issue the 8 WMMAs back-to-back so
    //      LDS latency overlaps the XDL pipe instead of serializing it. ----
    {
      const int lr = lane & 15;
      const int hf = lane >> 4;
      const int arow = wave * 16 + lr;
      // A 16x32 layout: lanes0-15 K=0..15, lanes16-31 K=16..31
      v8h alo = *(const v8h*)&sA[arow][hf * 16];
      v8h ahi = *(const v8h*)&sA[arow][hf * 16 + 8];
      v16h af = __builtin_shufflevector(alo, ahi,
          0,1,2,3,4,5,6,7,8,9,10,11,12,13,14,15);

      v8h blo[NT], bhi[NT];
      #pragma unroll
      for (int nt = 0; nt < NT; ++nt) {
        // B 32x16 layout: lanes0-15 K=0..15, lanes16-31 K=16..31, col = lr
        const _Float16* bp = &sBt[nt * 16 + lr][hf * 16];
        blo[nt] = *(const v8h*)bp;
        bhi[nt] = *(const v8h*)(bp + 8);
      }
      #pragma unroll
      for (int nt = 0; nt < NT; ++nt) {
        v16h bf = __builtin_shufflevector(blo[nt], bhi[nt],
            0,1,2,3,4,5,6,7,8,9,10,11,12,13,14,15);
        acc[nt] = __builtin_amdgcn_wmma_f32_16x16x32_f16(
            false, af, false, bf, (short)0, acc[nt], false, false);
      }
    }
    __syncthreads();
  }

  // ---- store: C/D 16x16 f32 layout ----
  {
    const int lr = lane & 15;
    const int hf = lane >> 4;
    #pragma unroll
    for (int nt = 0; nt < NT; ++nt) {
      #pragma unroll
      for (int vv = 0; vv < 8; ++vv) {
        int m = m0 + wave * 16 + vv + 8 * hf;
        int n = n0 + nt * 16 + lr;
        if (m < M && n < N) {
          float val = acc[nt][vv];
          if (flags & 1) val += bias[n];
          if (flags & 2) val += C[(size_t)m * ldc + n];
          C[(size_t)m * ldc + n] = val;
        }
      }
    }
  }
}

// ---------------------------------------------------------------------------
// Elementwise / normalization kernels (one wave per row for norms)
// ---------------------------------------------------------------------------
__global__ void k_layernorm(const float* __restrict__ x, const float* __restrict__ g,
                            const float* __restrict__ b, float* __restrict__ y,
                            int rows, int D) {
  int wave = threadIdx.x >> 5, lane = threadIdx.x & 31;
  int row = blockIdx.x * 8 + wave;
  if (row >= rows) return;
  const float* xr = x + (size_t)row * D;
  float s = 0.f, s2 = 0.f;
  for (int j = lane; j < D; j += 32) { float v = xr[j]; s += v; s2 += v * v; }
  #pragma unroll
  for (int m = 16; m > 0; m >>= 1) { s += __shfl_xor(s, m); s2 += __shfl_xor(s2, m); }
  float mu = s / D;
  float inv = rsqrtf(s2 / D - mu * mu + 1e-5f);
  float* yr = y + (size_t)row * D;
  for (int j = lane; j < D; j += 32) yr[j] = (xr[j] - mu) * inv * g[j] + b[j];
}

__global__ void k_rmsnorm(const float* __restrict__ x, const float* __restrict__ w,
                          float* __restrict__ y, int rows, int D) {
  int wave = threadIdx.x >> 5, lane = threadIdx.x & 31;
  int row = blockIdx.x * 8 + wave;
  if (row >= rows) return;
  const float* xr = x + (size_t)row * D;
  float s2 = 0.f;
  for (int j = lane; j < D; j += 32) { float v = xr[j]; s2 += v * v; }
  #pragma unroll
  for (int m = 16; m > 0; m >>= 1) s2 += __shfl_xor(s2, m);
  float inv = rsqrtf(s2 / D + 1e-5f);
  float* yr = y + (size_t)row * D;
  for (int j = lane; j < D; j += 32) yr[j] = xr[j] * inv * w[j];
}

__global__ void k_gelu(float* __restrict__ x, long n) {
  long i = (long)blockIdx.x * blockDim.x + threadIdx.x;
  if (i < n) x[i] = geluf(x[i]);
}

// xi = silu(causal depthwise conv(xz[:, :256])), xz layout (BL, 512)
__global__ void k_dwconv_silu(const float* __restrict__ xz, const float* __restrict__ w,
                              const float* __restrict__ cb, float* __restrict__ xi) {
  int idx = blockIdx.x * blockDim.x + threadIdx.x;  // < BL*DINNER
  int d = idx & (DINNER - 1);
  int m = idx / DINNER;
  int b = m / LL, l = m - b * LL;
  float s = cb[d];
  #pragma unroll
  for (int t = 0; t < DCONV; ++t) {
    int ll = l - (DCONV - 1) + t;
    if (ll >= 0) s += xz[(size_t)(b * LL + ll) * 512 + d] * w[d * DCONV + t];
  }
  xi[idx] = siluf(s);
}

__global__ void k_softplus(float* __restrict__ dl, const float* __restrict__ dtb) {
  int idx = blockIdx.x * blockDim.x + threadIdx.x;  // < BL*DINNER
  int d = idx & (DINNER - 1);
  float v = dl[idx] + dtb[d];
  dl[idx] = (v > 20.f) ? v : log1pf(expf(v));
}

// Diagonal SSM scan: 16 lanes (states) per (b,d) channel, sequential over L.
__global__ void k_ssm(const float* __restrict__ delta, const float* __restrict__ xi,
                      const float* __restrict__ dbc, const float* __restrict__ A_log,
                      const float* __restrict__ Dv, float* __restrict__ y) {
  int t  = blockIdx.x * blockDim.x + threadIdx.x;
  int s  = t & (DSTATE - 1);
  int ch = t >> 4;                         // 0 .. B*DINNER-1
  int b  = ch / DINNER, d = ch - b * DINNER;
  float Ads = -__expf(A_log[d * DSTATE + s]);
  float Dd  = Dv[d];
  float h = 0.f;
  for (int l = 0; l < LL; ++l) {
    int row = b * LL + l;
    float dt = delta[(size_t)row * DINNER + d];
    float xv = xi[(size_t)row * DINNER + d];
    float Bv = dbc[(size_t)row * FF + DTRANK + s];
    float Cv = dbc[(size_t)row * FF + DTRANK + DSTATE + s];
    h = __expf(dt * Ads) * h + (dt * xv) * Bv;
    float p = h * Cv;
    p += __shfl_xor(p, 1, 16);
    p += __shfl_xor(p, 2, 16);
    p += __shfl_xor(p, 4, 16);
    p += __shfl_xor(p, 8, 16);
    if (s == 0) y[(size_t)row * DINNER + d] = p + xv * Dd;
  }
}

__global__ void k_mul_silu(float* __restrict__ y, const float* __restrict__ xz) {
  int idx = blockIdx.x * blockDim.x + threadIdx.x;  // < BL*DINNER
  int d = idx & (DINNER - 1);
  int m = idx / DINNER;
  y[idx] *= siluf(xz[(size_t)m * 512 + DINNER + d]);
}

// ---- BatchNorm (training-mode stats over all BL rows, per channel) ----
__global__ void k_bn_stats(const float* __restrict__ X, float* __restrict__ st, int C) {
  int c = blockIdx.y * 256 + threadIdx.x;
  if (c >= C) return;
  int r0 = blockIdx.x * 128;
  float s = 0.f, s2 = 0.f;
  for (int r = r0; r < r0 + 128; ++r) {
    float v = X[(size_t)r * C + c];
    s += v; s2 += v * v;
  }
  atomicAdd(&st[c], s);
  atomicAdd(&st[C + c], s2);
}

__global__ void k_bn_apply(float* __restrict__ X, const float* __restrict__ st,
                           const float* __restrict__ g, const float* __restrict__ b,
                           int C, int act) {
  long n = (long)BL * C;
  long i = (long)blockIdx.x * blockDim.x + threadIdx.x;
  if (i >= n) return;
  int c = (int)(i % C);
  float mu  = st[c] * (1.0f / BL);
  float var = st[C + c] * (1.0f / BL) - mu * mu;
  float v = (X[i] - mu) * rsqrtf(var + 1e-5f) * g[c] + b[c];
  X[i] = act ? geluf(v) : v;
}

__global__ void k_add_gelu(float* __restrict__ a, const float* __restrict__ b, long n) {
  long i = (long)blockIdx.x * blockDim.x + threadIdx.x;
  if (i < n) a[i] = geluf(a[i] + b[i]);
}

__global__ void k_mean_pool(const float* __restrict__ X, float* __restrict__ feat) {
  int b = blockIdx.x;
  for (int c = threadIdx.x; c < 512; c += 256) {
    float s = 0.f;
    for (int l = 0; l < LL; ++l) s += X[(size_t)(b * LL + l) * 512 + c];
    feat[b * 512 + c] = s * (1.0f / LL);
  }
}

__global__ void k_head_out(const float* __restrict__ z, const float* __restrict__ w2,
                           const float* __restrict__ b2, float* __restrict__ out, int Do) {
  int idx = blockIdx.x * blockDim.x + threadIdx.x;
  if (idx >= BB * Do) return;
  int b = idx / Do, o = idx - b * Do;
  float s = b2[o];
  for (int j = 0; j < 256; ++j) s += z[b * 256 + j] * w2[j * Do + o];
  out[b * Do + o] = s;
}

// ---------------------------------------------------------------------------
// Host driver
// ---------------------------------------------------------------------------
extern "C" void kernel_launch(void* const* d_in, const int* in_sizes, int n_in,
                              void* d_out, int out_size, void* d_ws, size_t ws_size,
                              hipStream_t stream) {
  (void)in_sizes; (void)n_in; (void)out_size; (void)ws_size;
  auto IN = [&](int i) { return (const float*)d_in[i]; };
  float* ws  = (float*)d_ws;
  float* out = (float*)d_out;

  // workspace layout (floats); conv buffers alias the dead mamba region
  const size_t o_xn = 0;
  const size_t o_h  = o_xn + (size_t)BL * FF;
  const size_t o_hn = o_h  + (size_t)BL * EE;
  const size_t o_xz = o_hn + (size_t)BL * EE;
  const size_t o_xi = o_xz + (size_t)BL * 512;
  const size_t o_db = o_xi + (size_t)BL * DINNER;
  const size_t o_dl = o_db + (size_t)BL * FF;
  const size_t o_y  = o_dl + (size_t)BL * DINNER;
  const size_t o_cA = o_xz;                       // reuse after mamba stack
  const size_t o_cB = o_cA + (size_t)BL * 512;
  const size_t o_cS = o_cB + (size_t)BL * 512;
  const size_t o_st = o_cS + (size_t)BL * 512;
  const size_t o_ft = o_st + 1024;
  const size_t o_z  = o_ft + (size_t)BB * 512;

  float *xn = ws + o_xn, *h = ws + o_h, *hn = ws + o_hn;
  float *xz = ws + o_xz, *xi = ws + o_xi, *dbc = ws + o_db;
  float *dl = ws + o_dl, *yb = ws + o_y;
  float *cA = ws + o_cA, *cB = ws + o_cB, *cS = ws + o_cS;
  float *st = ws + o_st, *ft = ws + o_ft, *zb = ws + o_z;

  auto GEMM = [&](const float* A, int lda, const float* Bw, const float* bias,
                  float* C, int ldc, int M, int N, int K, int kind, int flags, int Cin) {
    dim3 g((N + BN - 1) / BN, (M + BM - 1) / BM);
    gemm_wmma_f16<<<g, dim3(128), 0, stream>>>(A, lda, Bw, bias, C, ldc,
                                               M, N, K, kind, flags, Cin, LL);
  };
  auto LN  = [&](const float* x, const float* g, const float* b, float* y, int rows, int D) {
    k_layernorm<<<(rows + 7) / 8, 256, 0, stream>>>(x, g, b, y, rows, D);
  };
  auto RMS = [&](const float* x, const float* w, float* y, int rows, int D) {
    k_rmsnorm<<<(rows + 7) / 8, 256, 0, stream>>>(x, w, y, rows, D);
  };
  auto GELU = [&](float* x, long n) {
    k_gelu<<<(int)((n + 255) / 256), 256, 0, stream>>>(x, n);
  };
  auto BNS = [&](const float* X, int C) {
    hipMemsetAsync((void*)st, 0, 1024 * sizeof(float), stream);
    dim3 g(BL / 128, (C + 255) / 256);
    k_bn_stats<<<g, 256, 0, stream>>>(X, st, C);
  };
  auto BNA = [&](float* X, const float* g, const float* b, int C, int act) {
    long n = (long)BL * C;
    k_bn_apply<<<(int)((n + 255) / 256), 256, 0, stream>>>(X, st, g, b, C, act);
  };

  // 1) input LN (F=40) + embedding GEMM + LN + GELU
  LN(IN(0), IN(1), IN(2), xn, BL, FF);
  GEMM(xn, FF, IN(3), IN(4), h, EE, BL, EE, FF, 0, /*bias*/1, 0);
  LN(h, IN(5), IN(6), h, BL, EE);
  GELU(h, (long)BL * EE);

  // 2) Mamba layers
  for (int l = 0; l < NLAYERS; ++l) {
    int p = 7 + l * 10;
    RMS(h, IN(p + 0), hn, BL, EE);
    GEMM(hn, EE, IN(p + 1), nullptr, xz, 512, BL, 512, EE, 0, 0, 0);
    k_dwconv_silu<<<(BL * DINNER) / 256, 256, 0, stream>>>(xz, IN(p + 2), IN(p + 3), xi);
    GEMM(xi, DINNER, IN(p + 4), nullptr, dbc, FF, BL, FF, DINNER, 0, 0, 0);
    GEMM(dbc, FF, IN(p + 5), nullptr, dl, DINNER, BL, DINNER, DTRANK, 0, 0, 0);
    k_softplus<<<(BL * DINNER) / 256, 256, 0, stream>>>(dl, IN(p + 6));
    k_ssm<<<(BB * DINNER * DSTATE) / 256, 256, 0, stream>>>(dl, xi, dbc,
                                                            IN(p + 7), IN(p + 8), yb);
    k_mul_silu<<<(BL * DINNER) / 256, 256, 0, stream>>>(yb, xz);
    GEMM(yb, DINNER, IN(p + 9), nullptr, h, EE, BL, EE, DINNER, 0, /*accum*/2, 0);
  }

  // 3) final RMS, then conv trunk in (B,L,C) layout (convs = im2col WMMA GEMMs)
  RMS(h, IN(47), hn, BL, EE);
  const float* cur = hn;
  int cin = EE;
  const int couts[3] = {256, 384, 512};
  for (int bk = 0; bk < 3; ++bk) {
    int p = 48 + bk * 12;
    int cout = couts[bk];
    // shortcut 1x1 + BN
    GEMM(cur, cin, IN(p + 8), IN(p + 9), cS, cout, BL, cout, cin, 2, 1, cin);
    BNS(cS, cout); BNA(cS, IN(p + 10), IN(p + 11), cout, 0);
    // conv1 (k=3) + BN + GELU
    GEMM(cur, cin, IN(p + 0), IN(p + 1), cA, cout, BL, cout, 3 * cin, 1, 1, cin);
    BNS(cA, cout); BNA(cA, IN(p + 2), IN(p + 3), cout, 1);
    // conv2 (k=3) + BN
    GEMM(cA, cout, IN(p + 4), IN(p + 5), cB, cout, BL, cout, 3 * cout, 1, 1, cout);
    BNS(cB, cout); BNA(cB, IN(p + 6), IN(p + 7), cout, 0);
    // residual add + GELU (in place into cB -> next input)
    k_add_gelu<<<(int)(((long)BL * cout + 255) / 256), 256, 0, stream>>>(
        cB, cS, (long)BL * cout);
    cur = cB; cin = cout;
  }

  // 4) global mean pool + heads
  k_mean_pool<<<BB, 256, 0, stream>>>(cB, ft);
  for (int hd = 0; hd < 2; ++hd) {
    int p = 84 + hd * 6;
    int Do = (hd == 0) ? 2 : 1;
    GEMM(ft, 512, IN(p + 0), IN(p + 1), zb, 256, BB, 256, 512, 0, 1, 0);
    LN(zb, IN(p + 2), IN(p + 3), zb, BB, 256);
    GELU(zb, (long)BB * 256);
    k_head_out<<<1, 32, 0, stream>>>(zb, IN(p + 4), IN(p + 5),
                                     out + (hd == 0 ? 0 : BB * 2), Do);
  }
}
